// DGCNN_cls_687194767479
// MI455X (gfx1250) — matrix operand order repaired and checked by reading, not compiled, not run on hardware
//
#include <hip/hip_runtime.h>

typedef __attribute__((ext_vector_type(16))) _Float16 v16h;
typedef __attribute__((ext_vector_type(8)))  float    v8f;

#define B_      8
#define NPTS    2048
#define KNN     40
#define EPS_    1e-5f
#define SLOPE_  0.2f
#define NEG_INF (-3.402823466e+38f)

union H16 { v16h v; _Float16 h[16]; };
union F8  { v8f  v; float    f[8];  };

static __device__ __forceinline__ float leaky_bn(float v, float g, float b) {
    float s = g * rsqrtf(1.0f + EPS_);
    v = v * s + b;
    return v >= 0.0f ? v : SLOPE_ * v;
}

// ---- fragment builders: unconditional float4 (b128) loads, f32->f16 in-register ----
// A-layout (16-bit A 16x32): lane<16 -> K {0..7,16..23}; lane>=16 -> K {8..15,24..31}.
// Per lane & K-tile this is two contiguous 8-float runs.
static __device__ __forceinline__ v16h build_a_frag(const float* __restrict__ pa, int hi, int kt) {
    const float* s0 = pa + kt * 32 + hi * 8;
    const float* s1 = s0 + 16;
    float4 q0 = *reinterpret_cast<const float4*>(s0);
    float4 q1 = *reinterpret_cast<const float4*>(s0 + 4);
    float4 q2 = *reinterpret_cast<const float4*>(s1);
    float4 q3 = *reinterpret_cast<const float4*>(s1 + 4);
    H16 a;
    a.h[0]  = (_Float16)q0.x; a.h[1]  = (_Float16)q0.y; a.h[2]  = (_Float16)q0.z; a.h[3]  = (_Float16)q0.w;
    a.h[4]  = (_Float16)q1.x; a.h[5]  = (_Float16)q1.y; a.h[6]  = (_Float16)q1.z; a.h[7]  = (_Float16)q1.w;
    a.h[8]  = (_Float16)q2.x; a.h[9]  = (_Float16)q2.y; a.h[10] = (_Float16)q2.z; a.h[11] = (_Float16)q2.w;
    a.h[12] = (_Float16)q3.x; a.h[13] = (_Float16)q3.y; a.h[14] = (_Float16)q3.z; a.h[15] = (_Float16)q3.w;
    return a.v;
}

// B-layout (16-bit B 32x16): n = lane&15, same K order as A: lane<16 -> K {0..7,16..23}; lane>=16 -> K {8..15,24..31}.
static __device__ __forceinline__ v16h build_b_frag(const float* __restrict__ pb, int hi, int kt) {
    const float* s = pb + kt * 32 + hi * 8;
    float4 q0 = *reinterpret_cast<const float4*>(s);
    float4 q1 = *reinterpret_cast<const float4*>(s + 4);
    float4 q2 = *reinterpret_cast<const float4*>(s + 16);
    float4 q3 = *reinterpret_cast<const float4*>(s + 20);
    H16 b;
    b.h[0]  = (_Float16)q0.x; b.h[1]  = (_Float16)q0.y; b.h[2]  = (_Float16)q0.z; b.h[3]  = (_Float16)q0.w;
    b.h[4]  = (_Float16)q1.x; b.h[5]  = (_Float16)q1.y; b.h[6]  = (_Float16)q1.z; b.h[7]  = (_Float16)q1.w;
    b.h[8]  = (_Float16)q2.x; b.h[9]  = (_Float16)q2.y; b.h[10] = (_Float16)q2.z; b.h[11] = (_Float16)q2.w;
    b.h[12] = (_Float16)q3.x; b.h[13] = (_Float16)q3.y; b.h[14] = (_Float16)q3.z; b.h[15] = (_Float16)q3.w;
    return b.v;
}

// ---------------- x (B,3,N) -> feat0p (B,N,32) zero-padded ----------------
__global__ void transpose_pad_kernel(const float* __restrict__ x, float* __restrict__ feat0p) {
    int t = blockIdx.x * blockDim.x + threadIdx.x;
    if (t >= B_ * NPTS * 32) return;
    int c = t & 31;
    int n = (t >> 5) % NPTS;
    int b = t / (32 * NPTS);
    feat0p[t] = (c < 3) ? x[((size_t)b * 3 + c) * NPTS + n] : 0.0f;
}

// ---------------- per-point squared norms ----------------
__global__ void sqnorm_kernel(const float* __restrict__ feat, int ld, int coff, int C,
                              float* __restrict__ xx) {
    int t = blockIdx.x * blockDim.x + threadIdx.x;
    if (t >= B_ * NPTS) return;
    const float* p = feat + (size_t)t * ld + coff;
    float s = 0.0f;
    for (int c = 0; c < C; ++c) { float v = p[c]; s += v * v; }
    xx[t] = s;
}

// ---------------- KNN: WMMA inner products + top-k select ----------------
// One wave per 16 rows; C (multiple of 32) is compile-time so all fragments
// live in registers and the WMMA chain fully unrolls.
template<int C>
__global__ __launch_bounds__(32)
void knn_wmma_kernel(const float* __restrict__ feat, int ld, int coff,
                     const float* __restrict__ xx, int* __restrict__ idx) {
    constexpr int NKT = C / 32;
    __shared__ float tile[16][516];
    int b    = blockIdx.y;
    int row0 = blockIdx.x * 16;
    int lane = threadIdx.x;
    int m    = lane & 15;
    int hi   = lane >> 4;

    const float* fb = feat + ((size_t)b * NPTS) * ld + coff;
    const float* pa = fb + (size_t)(row0 + m) * ld;

    v16h afr[NKT];
#pragma unroll
    for (int kt = 0; kt < NKT; ++kt) afr[kt] = build_a_frag(pa, hi, kt);

    float dbest[KNN]; int ibest[KNN];
#pragma unroll
    for (int t = 0; t < KNN; ++t) { dbest[t] = NEG_INF; ibest[t] = 0; }

    for (int chunk = 0; chunk < NPTS / 512; ++chunk) {
        int c0 = chunk * 512;
        for (int ct = 0; ct < 32; ++ct) {
            int col0 = c0 + ct * 16;
            const float* pb = fb + (size_t)(col0 + m) * ld;
            F8 acc;
#pragma unroll
            for (int r = 0; r < 8; ++r) acc.f[r] = 0.0f;
#pragma unroll
            for (int kt = 0; kt < NKT; ++kt) {
                v16h bfr = build_b_frag(pb, hi, kt);
                acc.v = __builtin_amdgcn_wmma_f32_16x16x32_f16(
                            false, afr[kt], false, bfr, (short)0, acc.v, false, false);
            }
            float xxc = xx[(size_t)b * NPTS + col0 + m];
#pragma unroll
            for (int r = 0; r < 8; ++r) {
                int mm = r + hi * 8;                 // C/D layout: M = r + 8*(lane/16)
                tile[mm][ct * 16 + m] = 2.0f * acc.f[r] - xxc;
            }
        }
        // single wave: per-wave in-order DS counters make write->read safe
        if (lane < 16) {
            for (int j = 0; j < 512; ++j) {
                float v = tile[lane][j];
                if (v > dbest[KNN - 1]) {
                    int p = KNN - 1;
                    while (p > 0 && dbest[p - 1] < v) {
                        dbest[p] = dbest[p - 1]; ibest[p] = ibest[p - 1]; --p;
                    }
                    dbest[p] = v; ibest[p] = c0 + j;
                }
            }
        }
    }
    if (lane < 16) {
        int* op = idx + ((size_t)b * NPTS + row0 + lane) * KNN;
        for (int t = 0; t < KNN; ++t) op[t] = ibest[t];
    }
}

// ---------------- weight prep ----------------
// out(O,Kpad) = w[o][off+k] for k<K else 0   (extract strided view + pad)
__global__ void wpad_kernel(const float* __restrict__ w, int ldw, int off, int K, int Kpad,
                            float* __restrict__ out, int total) {
    int t = blockIdx.x * blockDim.x + threadIdx.x;
    if (t >= total) return;
    int k = t % Kpad, o = t / Kpad;
    out[t] = (k < K) ? w[(size_t)o * ldw + off + k] : 0.0f;
}

// out(O,Kpad) = W2-W1 = w[o][C+k]-w[o][k] for k<C else 0
__global__ void wdiffpad_kernel(const float* __restrict__ w, int C, int Kpad,
                                float* __restrict__ out, int total) {
    int t = blockIdx.x * blockDim.x + threadIdx.x;
    if (t >= total) return;
    int k = t % Kpad, o = t / Kpad;
    out[t] = (k < C) ? (w[(size_t)o * 2 * C + C + k] - w[(size_t)o * 2 * C + k]) : 0.0f;
}

// ---------------- generic WMMA GEMM: out = A(B,N,K) @ W(O,K)^T ----------------
// K is compile-time (multiple of 32); one wave per 16x16 output tile.
template<int K>
__global__ __launch_bounds__(32)
void gemm_wmma_kernel(const float* __restrict__ A, int lda, int aoff,
                      const float* __restrict__ W, int ldw,
                      float* __restrict__ out, int ldo, int ooff,
                      const float* __restrict__ g, const float* __restrict__ bb,
                      int fuse_bn) {
    constexpr int NKT = K / 32;
    int b  = blockIdx.z;
    int m0 = blockIdx.x * 16;
    int o0 = blockIdx.y * 16;
    int lane = threadIdx.x, m = lane & 15, hi = lane >> 4;

    const float* pa = A + ((size_t)b * NPTS) * lda + aoff + (size_t)(m0 + m) * lda;
    const float* pb = W + (size_t)(o0 + m) * ldw;

    F8 acc;
#pragma unroll
    for (int r = 0; r < 8; ++r) acc.f[r] = 0.0f;

#pragma unroll
    for (int kt = 0; kt < NKT; ++kt) {
        v16h afr = build_a_frag(pa, hi, kt);
        v16h bfr = build_b_frag(pb, hi, kt);
        acc.v = __builtin_amdgcn_wmma_f32_16x16x32_f16(
                    false, afr, false, bfr, (short)0, acc.v, false, false);
    }
#pragma unroll
    for (int r = 0; r < 8; ++r) {
        int mm = m0 + r + hi * 8;
        int oo = o0 + m;
        float v = acc.f[r];
        if (fuse_bn) v = leaky_bn(v, g[oo], bb[oo]);
        out[((size_t)b * NPTS + mm) * ldo + ooff + oo] = v;
    }
}

// ---------------- gather-max + BN + leaky (max commutes with monotone epilogue) ----
__global__ void gather_max_kernel(const float* __restrict__ G1, const float* __restrict__ P,
                                  const int* __restrict__ idx,
                                  const float* __restrict__ g, const float* __restrict__ bb,
                                  float* __restrict__ out, int ldo, int ooff, int O) {
    int t = blockIdx.x * blockDim.x + threadIdx.x;
    if (t >= B_ * NPTS * O) return;
    int o = t % O;
    int n = (t / O) % NPTS;
    int b = t / (O * NPTS);
    const int*   ip = idx + ((size_t)b * NPTS + n) * KNN;
    const float* Gb = G1 + (size_t)b * NPTS * O;
    float mv = NEG_INF;
    for (int j = 0; j < KNN; ++j)
        mv = fmaxf(mv, Gb[(size_t)ip[j] * O + o]);
    float v = mv + P[((size_t)b * NPTS + n) * O + o];
    v = leaky_bn(v, g[o], bb[o]);
    out[((size_t)b * NPTS + n) * ldo + ooff + o] = v;
}

// ---------------- global max + mean pool over N: f5 (B,N,1024) -> x8 (B,2048) ----
__global__ void reduce_max_mean_kernel(const float* __restrict__ f5, float* __restrict__ x8) {
    int t = blockIdx.x * blockDim.x + threadIdx.x;
    if (t >= B_ * 1024) return;
    int e = t % 1024, b = t / 1024;
    const float* p = f5 + (size_t)b * NPTS * 1024 + e;
    float mx = NEG_INF, sm = 0.0f;
    for (int n = 0; n < NPTS; ++n) {
        float v = p[(size_t)n * 1024];
        mx = fmaxf(mx, v); sm += v;
    }
    x8[(size_t)b * 2048 + e]        = mx;
    x8[(size_t)b * 2048 + 1024 + e] = sm * (1.0f / (float)NPTS);
}

// ---------------- tiny FC head (B=8 rows) ----------------
__global__ void fc_kernel(const float* __restrict__ in, int K,
                          const float* __restrict__ W, const float* __restrict__ bias,
                          const float* __restrict__ g, const float* __restrict__ bb,
                          int act, float* __restrict__ out, int O) {
    int t = blockIdx.x * blockDim.x + threadIdx.x;
    if (t >= B_ * O) return;
    int o = t % O, b = t / O;
    const float* ip = in + (size_t)b * K;
    const float* wp = W  + (size_t)o * K;
    float s = 0.0f;
    for (int k = 0; k < K; ++k) s += ip[k] * wp[k];
    if (bias) s += bias[o];
    if (act)  s = leaky_bn(s, g[o], bb[o]);
    out[(size_t)b * O + o] = s;
}

// =====================================================================

extern "C" void kernel_launch(void* const* d_in, const int* in_sizes, int n_in,
                              void* d_out, int out_size, void* d_ws, size_t ws_size,
                              hipStream_t stream) {
    (void)in_sizes; (void)n_in; (void)out_size; (void)ws_size;

    const float* x   = (const float*)d_in[0];
    const float* w1  = (const float*)d_in[1];  const float* g1 = (const float*)d_in[2];  const float* b1 = (const float*)d_in[3];
    const float* w2  = (const float*)d_in[4];  const float* g2 = (const float*)d_in[5];  const float* b2 = (const float*)d_in[6];
    const float* w3  = (const float*)d_in[7];  const float* g3 = (const float*)d_in[8];  const float* b3 = (const float*)d_in[9];
    const float* w4  = (const float*)d_in[10]; const float* g4 = (const float*)d_in[11]; const float* b4 = (const float*)d_in[12];
    const float* w5  = (const float*)d_in[13]; const float* g5 = (const float*)d_in[14]; const float* b5 = (const float*)d_in[15];
    const float* wl1 = (const float*)d_in[16]; const float* g6 = (const float*)d_in[17]; const float* b6 = (const float*)d_in[18];
    const float* wl2 = (const float*)d_in[19]; const float* bl2 = (const float*)d_in[20];
    const float* g7  = (const float*)d_in[21]; const float* b7  = (const float*)d_in[22];
    const float* wl3 = (const float*)d_in[23]; const float* bl3 = (const float*)d_in[24];
    float* dout = (float*)d_out;

    // ---- carve workspace ----
    char*  ws  = (char*)d_ws;
    size_t off = 0;
    auto carve = [&](size_t bytes) -> void* {
        void* p = ws + off;
        off = (off + bytes + 255) & ~(size_t)255;
        return p;
    };
    float* feat0p = (float*)carve((size_t)B_ * NPTS * 32  * 4);   // padded (B,N,32)
    float* xx     = (float*)carve((size_t)B_ * NPTS       * 4);
    int*   idx    = (int*)  carve((size_t)B_ * NPTS * KNN * 4);
    float* G1     = (float*)carve((size_t)B_ * NPTS * 256 * 4);
    float* Pt     = (float*)carve((size_t)B_ * NPTS * 256 * 4);
    float* Wd     = (float*)carve((size_t)256 * 256       * 4);
    float* W1p    = (float*)carve((size_t)64 * 32         * 4);
    float* x5f    = (float*)carve((size_t)B_ * NPTS * 512 * 4);
    float* f5     = (float*)carve((size_t)B_ * NPTS * 1024* 4);
    float* x8     = (float*)carve((size_t)B_ * 2048       * 4);
    float* f8     = (float*)carve((size_t)B_ * 512        * 4);
    float* f9     = (float*)carve((size_t)B_ * 256        * 4);

    const int TB = 256;
    auto cdiv = [](int a, int b) { return (a + b - 1) / b; };

    // ---- x -> padded (B,N,32) ----
    transpose_pad_kernel<<<cdiv(B_ * NPTS * 32, TB), TB, 0, stream>>>(x, feat0p);

    // ================= layer 1 (C=3 padded to 32, O=64) =================
    sqnorm_kernel<<<cdiv(B_ * NPTS, TB), TB, 0, stream>>>(feat0p, 32, 0, 3, xx);
    knn_wmma_kernel<32><<<dim3(NPTS / 16, B_), 32, 0, stream>>>(feat0p, 32, 0, xx, idx);
    wpad_kernel<<<cdiv(64 * 32, TB), TB, 0, stream>>>(w1, 6, 0, 3, 32, W1p, 64 * 32);
    wdiffpad_kernel<<<cdiv(64 * 32, TB), TB, 0, stream>>>(w1, 3, 32, Wd, 64 * 32);
    gemm_wmma_kernel<32><<<dim3(NPTS / 16, 64 / 16, B_), 32, 0, stream>>>(
        feat0p, 32, 0, W1p, 32, G1, 64, 0, nullptr, nullptr, 0);
    gemm_wmma_kernel<32><<<dim3(NPTS / 16, 64 / 16, B_), 32, 0, stream>>>(
        feat0p, 32, 0, Wd, 32, Pt, 64, 0, nullptr, nullptr, 0);
    gather_max_kernel<<<cdiv(B_ * NPTS * 64, TB), TB, 0, stream>>>(
        G1, Pt, idx, g1, b1, x5f, 512, 0, 64);

    // ================= layer 2 (C=64, O=64) =================
    sqnorm_kernel<<<cdiv(B_ * NPTS, TB), TB, 0, stream>>>(x5f, 512, 0, 64, xx);
    knn_wmma_kernel<64><<<dim3(NPTS / 16, B_), 32, 0, stream>>>(x5f, 512, 0, xx, idx);
    wdiffpad_kernel<<<cdiv(64 * 64, TB), TB, 0, stream>>>(w2, 64, 64, Wd, 64 * 64);
    gemm_wmma_kernel<64><<<dim3(NPTS / 16, 64 / 16, B_), 32, 0, stream>>>(
        x5f, 512, 0, w2, 128, G1, 64, 0, nullptr, nullptr, 0);
    gemm_wmma_kernel<64><<<dim3(NPTS / 16, 64 / 16, B_), 32, 0, stream>>>(
        x5f, 512, 0, Wd, 64, Pt, 64, 0, nullptr, nullptr, 0);
    gather_max_kernel<<<cdiv(B_ * NPTS * 64, TB), TB, 0, stream>>>(
        G1, Pt, idx, g2, b2, x5f, 512, 64, 64);

    // ================= layer 3 (C=64, O=128) =================
    sqnorm_kernel<<<cdiv(B_ * NPTS, TB), TB, 0, stream>>>(x5f, 512, 64, 64, xx);
    knn_wmma_kernel<64><<<dim3(NPTS / 16, B_), 32, 0, stream>>>(x5f, 512, 64, xx, idx);
    wdiffpad_kernel<<<cdiv(128 * 64, TB), TB, 0, stream>>>(w3, 64, 64, Wd, 128 * 64);
    gemm_wmma_kernel<64><<<dim3(NPTS / 16, 128 / 16, B_), 32, 0, stream>>>(
        x5f, 512, 64, w3, 128, G1, 128, 0, nullptr, nullptr, 0);
    gemm_wmma_kernel<64><<<dim3(NPTS / 16, 128 / 16, B_), 32, 0, stream>>>(
        x5f, 512, 64, Wd, 64, Pt, 128, 0, nullptr, nullptr, 0);
    gather_max_kernel<<<cdiv(B_ * NPTS * 128, TB), TB, 0, stream>>>(
        G1, Pt, idx, g3, b3, x5f, 512, 128, 128);

    // ================= layer 4 (C=128, O=256) =================
    sqnorm_kernel<<<cdiv(B_ * NPTS, TB), TB, 0, stream>>>(x5f, 512, 128, 128, xx);
    knn_wmma_kernel<128><<<dim3(NPTS / 16, B_), 32, 0, stream>>>(x5f, 512, 128, xx, idx);
    wdiffpad_kernel<<<cdiv(256 * 128, TB), TB, 0, stream>>>(w4, 128, 128, Wd, 256 * 128);
    gemm_wmma_kernel<128><<<dim3(NPTS / 16, 256 / 16, B_), 32, 0, stream>>>(
        x5f, 512, 128, w4, 256, G1, 256, 0, nullptr, nullptr, 0);
    gemm_wmma_kernel<128><<<dim3(NPTS / 16, 256 / 16, B_), 32, 0, stream>>>(
        x5f, 512, 128, Wd, 128, Pt, 256, 0, nullptr, nullptr, 0);
    gather_max_kernel<<<cdiv(B_ * NPTS * 256, TB), TB, 0, stream>>>(
        G1, Pt, idx, g4, b4, x5f, 512, 256, 256);

    // ---- f5 = leaky(bn(x5 @ w5^T)) : (B,N,512) -> (B,N,1024) ----
    gemm_wmma_kernel<512><<<dim3(NPTS / 16, 1024 / 16, B_), 32, 0, stream>>>(
        x5f, 512, 0, w5, 512, f5, 1024, 0, g5, b5, 1);

    // ---- global pool -> x8 (B,2048) ----
    reduce_max_mean_kernel<<<cdiv(B_ * 1024, TB), TB, 0, stream>>>(f5, x8);

    // ---- head ----
    fc_kernel<<<cdiv(B_ * 512, TB), TB, 0, stream>>>(x8, 2048, wl1, nullptr, g6, b6, 1, f8, 512);
    fc_kernel<<<cdiv(B_ * 256, TB), TB, 0, stream>>>(f8, 512,  wl2, bl2,    g7, b7, 1, f9, 256);
    fc_kernel<<<cdiv(B_ * 40,  TB), TB, 0, stream>>>(f9, 256,  wl3, bl3, nullptr, nullptr, 0, dout, 40);
}
